// MultiTokenConceptLayer_25039659336372
// MI455X (gfx1250) — matrix operand order repaired and checked by reading, not compiled, not run on hardware
//
#include <hip/hip_runtime.h>
#include <stdint.h>

// ---------------------------------------------------------------------------
// MultiTokenConceptLayer for MI455X (gfx1250, wave32, WMMA)
//
// GEMM (16384x1024x1024, f32 in/out) on the bf16 matrix pipe via
// split-precision: x = hi(bf16,rne) + lo(bf16), D += Ahi*Bhi + Ahi*Blo + Alo*Bhi
// -> 3x v_wmma_f32_16x16x32_bf16 per tile = 8/3x the K-throughput of the
// native V_WMMA_F32_16X16X4_F32 path at ~2^-16 relative accuracy.
//
// Round 2: wave tile 32x64 (2x4 WMMA tiles), block tile 128x128 -> 24 WMMAs
// per 24 ds_load_b128 per K-step (was 12:16), branchless K-tile prefetch.
// ---------------------------------------------------------------------------

typedef __attribute__((ext_vector_type(16))) __bf16    v16bf;
typedef __attribute__((ext_vector_type(8)))  float     v8f;
typedef __attribute__((ext_vector_type(4)))  float     v4f;
typedef __attribute__((ext_vector_type(4)))  uint32_t  v4u;
typedef __attribute__((ext_vector_type(2)))  uint32_t  v2u;

#define D_DIM   1024
#define S_DIM   2048
#define B_DIM   8
#define Q_DIM   16
#define C_DIM   64
#define T_DIM   4
#define OUT_S   (S_DIM + C_DIM * T_DIM)   // 2304 rows per batch
#define M_DIM   (B_DIM * S_DIM)           // 16384 GEMM rows

#define BM  128
#define BN  128
#define BK  32
#define LDA 40    // padded LDS row stride in halfwords (80 B, multiple of 16 B)

union Frag { v16bf v; v4u q[2]; };

// round-to-nearest-even truncation of f32 to bf16 bit pattern (in high half)
__device__ __forceinline__ uint32_t bf16_hi_rne(uint32_t u) {
  return (u + 0x7FFFu + ((u >> 16) & 1u)) & 0xFFFF0000u;
}

__device__ __forceinline__ void split2(float x, uint32_t& hi, uint32_t& lo) {
  uint32_t u  = __float_as_uint(x);
  uint32_t hb = bf16_hi_rne(u);
  float    r  = x - __uint_as_float(hb);      // exact residual
  hi = hb >> 16;
  lo = __float_as_uint(r) >> 16;              // truncate residual to bf16
}

__global__ __launch_bounds__(256, 2)
void mtcl_gemm_bias_kernel(const float* __restrict__ A,     // [16384,1024] hidden
                           const float* __restrict__ W,     // [1024,1024]
                           const float* __restrict__ bias,  // [1024]
                           float* __restrict__ out)         // [8,2304,1024]
{
  __shared__ unsigned short sAhi[BM * LDA];
  __shared__ unsigned short sAlo[BM * LDA];
  __shared__ unsigned short sBhi[BN * LDA];
  __shared__ unsigned short sBlo[BN * LDA];

  const int tid  = threadIdx.x;
  const int lane = tid & 31;
  const int hlf  = lane >> 4;      // which 16-lane half of the wave
  const int l    = lane & 15;
  const int wave = tid >> 5;       // 8 waves: 4 along M x 2 along N
  const int waveM = (wave & 3) * 32;
  const int waveN = (wave >> 2) * 64;

  const int mBase = blockIdx.y * BM;
  const int nBase = blockIdx.x * BN;

  v8f acc[2][4] = {};

  for (int kb = 0; kb < D_DIM; kb += BK) {
    // branchless prefetch base: clamp at the last tile (re-prefetch is harmless)
    const int kpf = (kb + BK < D_DIM) ? (kb + BK) : kb;

    // ---- global f32 -> LDS bf16 hi/lo planes (A: 128x32, B: 128x32) ----
    #pragma unroll
    for (int i = 0; i < 4; ++i) {
      int idx = tid + i * 256;               // 0..1023
      int r   = idx >> 3;
      int c4  = (idx & 7) << 2;
      const float* gp = &A[(size_t)(mBase + r) * D_DIM + kb + c4];
      v4f f = *(const v4f*)gp;
      __builtin_prefetch(&A[(size_t)(mBase + r) * D_DIM + kpf + c4], 0, 1);
      uint32_t h0,l0,h1,l1,h2,l2,h3,l3;
      split2(f.x,h0,l0); split2(f.y,h1,l1); split2(f.z,h2,l2); split2(f.w,h3,l3);
      v2u hp; hp.x = h0 | (h1 << 16); hp.y = h2 | (h3 << 16);
      v2u lp; lp.x = l0 | (l1 << 16); lp.y = l2 | (l3 << 16);
      *(v2u*)&sAhi[r * LDA + c4] = hp;
      *(v2u*)&sAlo[r * LDA + c4] = lp;
    }
    #pragma unroll
    for (int i = 0; i < 4; ++i) {
      int idx = tid + i * 256;               // 0..1023
      int r   = idx >> 3;
      int c4  = (idx & 7) << 2;
      const float* gp = &W[(size_t)(nBase + r) * D_DIM + kb + c4];
      v4f f = *(const v4f*)gp;
      __builtin_prefetch(&W[(size_t)(nBase + r) * D_DIM + kpf + c4], 0, 1);
      uint32_t h0,l0,h1,l1,h2,l2,h3,l3;
      split2(f.x,h0,l0); split2(f.y,h1,l1); split2(f.z,h2,l2); split2(f.w,h3,l3);
      v2u hp; hp.x = h0 | (h1 << 16); hp.y = h2 | (h3 << 16);
      v2u lp; lp.x = l0 | (l1 << 16); lp.y = l2 | (l3 << 16);
      *(v2u*)&sBhi[r * LDA + c4] = hp;
      *(v2u*)&sBlo[r * LDA + c4] = lp;
    }
    __syncthreads();

    // ---- build WMMA fragments per ISA VGPR layout (2x ds_load_b128 each) ----
    // A 16x32 bf16: lane half h -> VGPR0-3 = K[8h..8h+7], VGPR4-7 = K[16+8h..+7]
    Frag aHi[2], aLo[2], bHi[4], bLo[4];
    #pragma unroll
    for (int mi = 0; mi < 2; ++mi) {
      int row = waveM + mi * 16 + l;
      const unsigned short* p0 = &sAhi[row * LDA + hlf * 8];
      const unsigned short* p1 = &sAlo[row * LDA + hlf * 8];
      aHi[mi].q[0] = *(const v4u*)p0;  aHi[mi].q[1] = *(const v4u*)(p0 + 16);
      aLo[mi].q[0] = *(const v4u*)p1;  aLo[mi].q[1] = *(const v4u*)(p1 + 16);
    }
    // B 32x16 bf16: lane holds column n=l; half h -> K[8h..8h+7], K[16+8h..+7]
    #pragma unroll
    for (int ni = 0; ni < 4; ++ni) {
      int row = waveN + ni * 16 + l;
      const unsigned short* p0 = &sBhi[row * LDA + hlf * 8];
      const unsigned short* p1 = &sBlo[row * LDA + hlf * 8];
      bHi[ni].q[0] = *(const v4u*)p0;  bHi[ni].q[1] = *(const v4u*)(p0 + 16);
      bLo[ni].q[0] = *(const v4u*)p1;  bLo[ni].q[1] = *(const v4u*)(p1 + 16);
    }

    // ---- 2x4 tiles x 3 split-precision WMMAs = 24 wmma per K-step per wave ----
    #pragma unroll
    for (int mi = 0; mi < 2; ++mi) {
      #pragma unroll
      for (int ni = 0; ni < 4; ++ni) {
        acc[mi][ni] = __builtin_amdgcn_wmma_f32_16x16x32_bf16(
            false, aHi[mi].v, false, bHi[ni].v, (short)0, acc[mi][ni], false, false);
        acc[mi][ni] = __builtin_amdgcn_wmma_f32_16x16x32_bf16(
            false, aHi[mi].v, false, bLo[ni].v, (short)0, acc[mi][ni], false, false);
        acc[mi][ni] = __builtin_amdgcn_wmma_f32_16x16x32_bf16(
            false, aLo[mi].v, false, bHi[ni].v, (short)0, acc[mi][ni], false, false);
      }
    }
    __syncthreads();
  }

  // ---- epilogue: D layout VGPR i -> row (i + 8*half), col = lane%16 ----
  #pragma unroll
  for (int ni = 0; ni < 4; ++ni) {
    int col = nBase + waveN + ni * 16 + l;
    float bv = bias[col];
    #pragma unroll
    for (int mi = 0; mi < 2; ++mi) {
      #pragma unroll
      for (int i = 0; i < 8; ++i) {
        int m  = mBase + waveM + mi * 16 + i + hlf * 8;
        int bb = m >> 11;                 // m / 2048
        int s  = m & (S_DIM - 1);
        out[((size_t)bb * OUT_S + s) * D_DIM + col] = acc[mi][ni][i] + bv;
      }
    }
  }
}

// ---------------------------------------------------------------------------
// Concept routing + token emission. One block per batch element; ~8 MB output,
// negligible next to the GEMM.
// ---------------------------------------------------------------------------
__global__ __launch_bounds__(256)
void mtcl_tokens_kernel(const float* __restrict__ csig,    // [8,16,1024]
                        const float* __restrict__ keys,    // [64,1024]
                        const float* __restrict__ values,  // [64,4,1024]
                        float* __restrict__ out)           // [8,2304,1024]
{
  const int b   = blockIdx.x;
  const int tid = threadIdx.x;

  __shared__ float dists[Q_DIM * C_DIM];
  __shared__ float smallest[Q_DIM];
  __shared__ int   chosen[Q_DIM];
  __shared__ int   kcount[C_DIM];        // tokens kept per concept (0 = unselected)
  __shared__ float vInv[C_DIM * T_DIM];  // 1/||values[c,t]||

  // cosine distances (norms folded into the same pass over D)
  for (int p = tid; p < Q_DIM * C_DIM; p += 256) {
    int q = p >> 6, c = p & 63;
    const float* qp = csig + ((size_t)b * Q_DIM + q) * D_DIM;
    const float* kp = keys + (size_t)c * D_DIM;
    float dot = 0.f, qq = 0.f, kk = 0.f;
    #pragma unroll 4
    for (int d = 0; d < D_DIM; ++d) {
      float a = qp[d], e = kp[d];
      dot = fmaf(a, e, dot); qq = fmaf(a, a, qq); kk = fmaf(e, e, kk);
    }
    float den = fmaxf(sqrtf(qq), 1e-12f) * fmaxf(sqrtf(kk), 1e-12f);
    dists[p] = 1.0f - dot / den;
  }
  // value-row inverse norms: 256 (c,t) pairs, one per thread
  {
    const float* vp = values + (size_t)tid * D_DIM;
    float s = 0.f;
    #pragma unroll 4
    for (int d = 0; d < D_DIM; ++d) s = fmaf(vp[d], vp[d], s);
    vInv[tid] = 1.0f / fmaxf(sqrtf(s), 1e-12f);
  }
  __syncthreads();

  // per-query min + argmin (first-hit tie-break matches jnp.argmin)
  if (tid < Q_DIM) {
    float best = dists[tid * C_DIM]; int bi = 0;
    for (int c = 1; c < C_DIM; ++c) {
      float v = dists[tid * C_DIM + c];
      if (v < best) { best = v; bi = c; }
    }
    smallest[tid] = best; chosen[tid] = bi;
  }
  __syncthreads();

  // per-concept: first selecting query -> gate -> token count
  if (tid < C_DIM) {
    int fp = Q_DIM;
    for (int q = Q_DIM - 1; q >= 0; --q)
      if (smallest[q] <= 1.0f && chosen[q] == tid) fp = q;
    int kc = 0;
    if (fp < Q_DIM) {
      float g = fminf(fmaxf(1.0f - smallest[fp], 0.0f), 1.0f);
      kc = (int)ceilf(g * (float)T_DIM);
      if (kc < 1) kc = 1;
    }
    kcount[tid] = kc;
  }
  __syncthreads();

  // emit tokens: rows [2048, 2304) of this batch, float4-coalesced
  for (int idx = tid; idx < C_DIM * T_DIM * (D_DIM / 4); idx += 256) {
    int d4 = (idx & 255) << 2;
    int ct = idx >> 8;           // 0..255 = c*4 + t
    int c  = ct >> 2, t = ct & 3;
    v4f o = {0.f, 0.f, 0.f, 0.f};
    if (t < kcount[c]) {
      float inv = vInv[ct];
      v4f v = *(const v4f*)&values[(size_t)ct * D_DIM + d4];
      o.x = v.x * inv; o.y = v.y * inv; o.z = v.z * inv; o.w = v.w * inv;
    }
    *(v4f*)&out[((size_t)b * OUT_S + S_DIM + ct) * D_DIM + d4] = o;
  }
}

extern "C" void kernel_launch(void* const* d_in, const int* in_sizes, int n_in,
                              void* d_out, int out_size, void* d_ws, size_t ws_size,
                              hipStream_t stream) {
  (void)in_sizes; (void)n_in; (void)out_size; (void)d_ws; (void)ws_size;
  const float* hidden = (const float*)d_in[0];  // [8,2048,1024]
  const float* csig   = (const float*)d_in[1];  // [8,16,1024]
  const float* keys   = (const float*)d_in[2];  // [64,1024]
  const float* values = (const float*)d_in[3];  // [64,4,1024]
  const float* W      = (const float*)d_in[4];  // [1024,1024]
  const float* bias   = (const float*)d_in[5];  // [1024]
  float* out = (float*)d_out;                   // [8,2304,1024]

  dim3 grid(D_DIM / BN, M_DIM / BM);            // (8, 128)
  mtcl_gemm_bias_kernel<<<grid, 256, 0, stream>>>(hidden, W, bias, out);
  mtcl_tokens_kernel<<<B_DIM, 256, 0, stream>>>(csig, keys, values, out);
}
